// WalkConv_29480655519882
// MI455X (gfx1250) — matrix operand order repaired and checked by reading, not compiled, not run on hardware
//
#include <hip/hip_runtime.h>
#include <hip/hip_bf16.h>
#include <stdint.h>

// ---------------------------------------------------------------------------
// WalkConv for MI455X (gfx1250), f16 WMMA path (f32 accumulate).
//   mtr1 = (mtr.reshape(768,6144) @ F) / 8, F dense f16, block-swizzled so
//   every WMMA B fragment is one contiguous 32B load per lane.
// MLP weights are pre-packed (f16, zero-padded) in B-fragment order; MLP
// activations are staged through LDS so A fragments are unconditional reads.
// ---------------------------------------------------------------------------

typedef __attribute__((ext_vector_type(16))) _Float16 v16h;
typedef __attribute__((ext_vector_type(8)))  _Float16 v8h;
typedef __attribute__((ext_vector_type(4)))  _Float16 v4h;
typedef __attribute__((ext_vector_type(8)))  float    v8f;

#define NN        768
#define HH        8
#define KDIM      6144          // NN*HH
#define EDGES     24576
#define EDGE_DIM  4
#define NODE_DIM  8

// Packed-weight block indices (each block = 32K x 16N = 512 f16 = 1KB).
#define EW0_B 0    // 1 Kchunk x 2 Ntiles
#define EW1_B 2
#define EW2_B 4
#define EW3_B 6    // 1 x 4
#define OW0_B 10   // 2 Kchunks x 2 Ntiles, order [kc][nt]
#define OW1_B 14
#define OW2_B 16
#define OW3_B 18   // 1 x 1
#define WP_BLOCKS 19

static __device__ __forceinline__ v8f wmma16(v16h a, v16h b, v8f c) {
  return __builtin_amdgcn_wmma_f32_16x16x32_f16(false, a, false, b, (short)0, c,
                                                false, false);
}

// Packed B fragment: contiguous 32B per lane.
static __device__ __forceinline__ v16h bfrag_p(const _Float16* __restrict__ P,
                                               int blk, int lane) {
  return *(const v16h*)(P + blk * 512 + lane * 16);
}

// Store a 16x16 f32 D tile (+bias, optional ReLU) into LDS staging.
static __device__ __forceinline__ void dstore(float* S, int stride, v8f c,
                                              int n0, const float* __restrict__ bias,
                                              bool relu, int lane) {
  int n  = n0 + (lane & 15);
  int mb = 8 * (lane >> 4);
  float b = bias[n];
#pragma unroll
  for (int v = 0; v < 8; ++v) {
    float x = c[v] + b;
    if (relu) x = fmaxf(x, 0.0f);
    S[(mb + v) * stride + n] = x;
  }
}

// Rebuild a 16x32 f16 A fragment from LDS (columns k0..k0+31, all pre-filled).
static __device__ __forceinline__ v16h afrag_lds(const float* S, int stride,
                                                 int k0, int lane) {
  int m = lane & 15, kh = lane >> 4;
  v16h a;
#pragma unroll
  for (int e = 0; e < 16; ++e) {
    int kl = kh * 16 + e;
    a[e] = (_Float16)S[m * stride + k0 + kl];
  }
  return a;
}

// --------------------------- prep kernels ----------------------------------

__global__ void __launch_bounds__(256)
f32_to_f16_4(const float4* __restrict__ src, v4h* __restrict__ dst, int n4) {
  int i = blockIdx.x * 256 + threadIdx.x;
  if (i >= n4) return;
  float4 v = src[i];
  v4h h;
  h[0] = (_Float16)v.x; h[1] = (_Float16)v.y;
  h[2] = (_Float16)v.z; h[3] = (_Float16)v.w;
  dst[i] = h;
}

__global__ void __launch_bounds__(256)
zero16B(uint4* __restrict__ p, int n16) {
  int i = blockIdx.x * 256 + threadIdx.x;
  if (i < n16) p[i] = make_uint4(0u, 0u, 0u, 0u);
}

// Pack W[out_dim][in_dim] f32 into f16 B-fragment order, zero padded.
// Blocks ordered [kchunk][ntile]; within a block: [lane][e] contiguous.
__global__ void __launch_bounds__(256)
pack_w(const float* __restrict__ W, int in_dim, int out_dim, int ntiles,
       _Float16* __restrict__ dst, int total) {
  int d = blockIdx.x * 256 + threadIdx.x;
  if (d >= total) return;
  int blk = d >> 9;
  int kc = blk / ntiles, nt = blk % ntiles;
  int r = d & 511;
  int lane = r >> 4, e = r & 15;
  int n = nt * 16 + (lane & 15);
  int k = kc * 32 + (lane >> 4) * 16 + e;
  float v = (n < out_dim && k < in_dim) ? W[n * in_dim + k] : 0.0f;
  dst[d] = (_Float16)v;
}

// --------------------------- edge MLP + scatter ----------------------------
// One wave per 16 edges: 20 -> 32 -> 32 -> 32 -> 64, scattered as 8x8 f16
// blocks into the block-swizzled dense F at (src*8+c, dst*8+t).

__global__ void __launch_bounds__(256)
edge_mlp_scatter(const float* __restrict__ edge_attr,
                 const float* __restrict__ feat,
                 const long long* __restrict__ ei,   // [2][E]
                 const _Float16* __restrict__ Wp,
                 const float* __restrict__ eb0, const float* __restrict__ eb1,
                 const float* __restrict__ eb2, const float* __restrict__ eb3,
                 _Float16* __restrict__ F16) {
  __shared__ float S[8][16 * 72];
  const int wave = threadIdx.x >> 5, lane = threadIdx.x & 31;
  float* st = S[wave];
  const int tile = blockIdx.x * 8 + wave;
  const int e0 = tile * 16;

  // Stage edge_in [16 rows x 32 cols] (20 real + zero pad): lane = column.
  {
    const int q = lane;
#pragma unroll
    for (int r = 0; r < 16; ++r) {
      float v = 0.0f;
      if (q < EDGE_DIM) {
        v = edge_attr[(e0 + r) * EDGE_DIM + q];
      } else if (q < EDGE_DIM + NODE_DIM) {
        long long s = ei[e0 + r];
        v = feat[s * NODE_DIM + (q - EDGE_DIM)];
      } else if (q < EDGE_DIM + 2 * NODE_DIM) {
        long long dnode = ei[EDGES + e0 + r];
        v = feat[dnode * NODE_DIM + (q - EDGE_DIM - NODE_DIM)];
      }
      st[r * 72 + q] = v;
    }
  }
  __syncthreads();
  v16h a = afrag_lds(st, 72, 0, lane);
  __syncthreads();

  v8f z = {};
  // L0: 20 -> 32
  dstore(st, 72, wmma16(a, bfrag_p(Wp, EW0_B + 0, lane), z), 0, eb0, true, lane);
  dstore(st, 72, wmma16(a, bfrag_p(Wp, EW0_B + 1, lane), z), 16, eb0, true, lane);
  __syncthreads();
  a = afrag_lds(st, 72, 0, lane);
  __syncthreads();
  // L1: 32 -> 32
  dstore(st, 72, wmma16(a, bfrag_p(Wp, EW1_B + 0, lane), z), 0, eb1, true, lane);
  dstore(st, 72, wmma16(a, bfrag_p(Wp, EW1_B + 1, lane), z), 16, eb1, true, lane);
  __syncthreads();
  a = afrag_lds(st, 72, 0, lane);
  __syncthreads();
  // L2: 32 -> 32
  dstore(st, 72, wmma16(a, bfrag_p(Wp, EW2_B + 0, lane), z), 0, eb2, true, lane);
  dstore(st, 72, wmma16(a, bfrag_p(Wp, EW2_B + 1, lane), z), 16, eb2, true, lane);
  __syncthreads();
  a = afrag_lds(st, 72, 0, lane);
  __syncthreads();
  // L3: 32 -> 64 (no ReLU)
#pragma unroll
  for (int t = 0; t < 4; ++t)
    dstore(st, 72, wmma16(a, bfrag_p(Wp, EW3_B + t, lane), z), t * 16, eb3,
           false, lane);
  __syncthreads();

  // Scatter 16x64 f16 into swizzled F: K index kk=src*8+c, N index nn=dst*8+t.
#pragma unroll
  for (int t = 0; t < 32; ++t) {
    int idx = lane + 32 * t;
    int r = idx >> 6, col = idx & 63;
    int c8 = col >> 3, t8 = col & 7;
    long long j = ei[e0 + r];
    long long k = ei[EDGES + e0 + r];
    int kk = (int)j * HH + c8;
    int nn = (int)k * HH + t8;
    int kb = kk >> 5, khh = (kk >> 4) & 1, eelem = kk & 15;
    int nb = nn >> 4, nl = nn & 15;
    size_t addr = (((size_t)kb * (KDIM / 16) + nb) * 32 + (khh * 16 + nl)) * 16 + eelem;
    F16[addr] = (_Float16)st[r * 72 + col];
  }
}

// --------------------------- walk contraction GEMM -------------------------
// One wave computes a 32x64 tile of mtr1 = A16 @ F16 * 0.125, K = 6144.
// F (75MB) is fully L2-resident (192MB L2), so B re-reads replay from L2.

__global__ void __launch_bounds__(256)
walk_gemm(const _Float16* __restrict__ A16, const _Float16* __restrict__ F16,
          float* __restrict__ M1) {
  const int wave = threadIdx.x >> 5, lane = threadIdx.x & 31;
  const int strip = blockIdx.x * 8 + wave;     // 24 * 96 strips
  const int i0 = (strip / 96) * 32;
  const int n0 = (strip % 96) * 64;
  const int m = lane & 15, kh = lane >> 4;

  v8f acc00 = {}, acc01 = {}, acc02 = {}, acc03 = {};
  v8f acc10 = {}, acc11 = {}, acc12 = {}, acc13 = {};
  const _Float16* Arow0 = A16 + (size_t)(i0 + m) * KDIM;
  const _Float16* Arow1 = A16 + (size_t)(i0 + 16 + m) * KDIM;
  const int nb0 = n0 >> 4;

  for (int kb = 0; kb < KDIM / 32; ++kb) {
    const int k0 = kb * 32;
    v8h lo0 = *(const v8h*)(Arow0 + k0 + kh * 16);
    v8h hi0 = *(const v8h*)(Arow0 + k0 + kh * 16 + 8);
    v8h lo1 = *(const v8h*)(Arow1 + k0 + kh * 16);
    v8h hi1 = *(const v8h*)(Arow1 + k0 + kh * 16 + 8);
    v16h a0 = __builtin_shufflevector(lo0, hi0, 0, 1, 2, 3, 4, 5, 6, 7,
                                      8, 9, 10, 11, 12, 13, 14, 15);
    v16h a1 = __builtin_shufflevector(lo1, hi1, 0, 1, 2, 3, 4, 5, 6, 7,
                                      8, 9, 10, 11, 12, 13, 14, 15);
    const _Float16* fb = F16 + (((size_t)kb * (KDIM / 16) + nb0) * 32 + lane) * 16;
    __builtin_prefetch((const void*)(fb + (size_t)(KDIM / 16) * 512), 0, 1);
    v16h b0 = *(const v16h*)(fb);
    v16h b1 = *(const v16h*)(fb + 512);
    v16h b2 = *(const v16h*)(fb + 1024);
    v16h b3 = *(const v16h*)(fb + 1536);
    acc00 = wmma16(a0, b0, acc00);
    acc01 = wmma16(a0, b1, acc01);
    acc02 = wmma16(a0, b2, acc02);
    acc03 = wmma16(a0, b3, acc03);
    acc10 = wmma16(a1, b0, acc10);
    acc11 = wmma16(a1, b1, acc11);
    acc12 = wmma16(a1, b2, acc12);
    acc13 = wmma16(a1, b3, acc13);
  }

  float* out0 = M1 + (size_t)(i0 + 8 * kh) * KDIM + n0 + m;
  float* out1 = M1 + (size_t)(i0 + 16 + 8 * kh) * KDIM + n0 + m;
#pragma unroll
  for (int v = 0; v < 8; ++v) {
    out0[(size_t)v * KDIM + 0]  = acc00[v] * 0.125f;
    out0[(size_t)v * KDIM + 16] = acc01[v] * 0.125f;
    out0[(size_t)v * KDIM + 32] = acc02[v] * 0.125f;
    out0[(size_t)v * KDIM + 48] = acc03[v] * 0.125f;
    out1[(size_t)v * KDIM + 0]  = acc10[v] * 0.125f;
    out1[(size_t)v * KDIM + 16] = acc11[v] * 0.125f;
    out1[(size_t)v * KDIM + 32] = acc12[v] * 0.125f;
    out1[(size_t)v * KDIM + 48] = acc13[v] * 0.125f;
  }
}

// --------------------------- output MLP ------------------------------------
// cat[i,k] = [mtr1[i,k,:], mtr1[k,i,:], feat[k], feat[i], (i==k)]  (33 wide)
// One wave per (i, 16 consecutive k): 33->32->32->32->8, out = mtr + y.

__global__ void __launch_bounds__(256)
out_mlp(const float* __restrict__ M1, const float* __restrict__ feat,
        const float* __restrict__ mtr, const _Float16* __restrict__ Wp,
        const float* __restrict__ ob0, const float* __restrict__ ob1,
        const float* __restrict__ ob2, const float* __restrict__ ob3,
        float* __restrict__ out) {
  __shared__ float S[8][16 * 72];
  const int wave = threadIdx.x >> 5, lane = threadIdx.x & 31;
  float* st = S[wave];
  const int tile = blockIdx.x * 8 + wave;      // 768 * 48 tiles
  const int i = tile / 48;
  const int k0 = (tile % 48) * 16;
  const int kh = lane >> 4;

  // Stage cat [16 rows x 64 cols] (33 real + zero pad), coalesced fill.
#pragma unroll
  for (int t = 0; t < 32; ++t) {
    int col = lane + 32 * (t & 1);
    int r = t >> 1;
    int krow = k0 + r;
    float v = 0.0f;
    if (col < 8)        v = M1[(size_t)i * KDIM + krow * 8 + col];
    else if (col < 16)  v = M1[(size_t)krow * KDIM + i * 8 + (col - 8)];
    else if (col < 24)  v = feat[krow * 8 + (col - 16)];
    else if (col < 32)  v = feat[i * 8 + (col - 24)];
    else if (col == 32) v = (i == krow) ? 1.0f : 0.0f;
    st[r * 72 + col] = v;
  }
  __syncthreads();
  v16h a0 = afrag_lds(st, 72, 0, lane);
  v16h a1 = afrag_lds(st, 72, 32, lane);
  __syncthreads();

  v8f z = {};
  // L0: 33 -> 32 (two K chunks accumulated)
  v8f c0 = wmma16(a0, bfrag_p(Wp, OW0_B + 0, lane), z);
  c0     = wmma16(a1, bfrag_p(Wp, OW0_B + 2, lane), c0);
  v8f c1 = wmma16(a0, bfrag_p(Wp, OW0_B + 1, lane), z);
  c1     = wmma16(a1, bfrag_p(Wp, OW0_B + 3, lane), c1);
  dstore(st, 72, c0, 0, ob0, true, lane);
  dstore(st, 72, c1, 16, ob0, true, lane);
  __syncthreads();
  v16h a = afrag_lds(st, 72, 0, lane);
  __syncthreads();
  // L1: 32 -> 32
  c0 = wmma16(a, bfrag_p(Wp, OW1_B + 0, lane), z);
  c1 = wmma16(a, bfrag_p(Wp, OW1_B + 1, lane), z);
  dstore(st, 72, c0, 0, ob1, true, lane);
  dstore(st, 72, c1, 16, ob1, true, lane);
  __syncthreads();
  a = afrag_lds(st, 72, 0, lane);
  __syncthreads();
  // L2: 32 -> 32
  c0 = wmma16(a, bfrag_p(Wp, OW2_B + 0, lane), z);
  c1 = wmma16(a, bfrag_p(Wp, OW2_B + 1, lane), z);
  dstore(st, 72, c0, 0, ob2, true, lane);
  dstore(st, 72, c1, 16, ob2, true, lane);
  __syncthreads();
  a = afrag_lds(st, 72, 0, lane);
  __syncthreads();
  // L3: 32 -> 8 (weights padded to 16 outs; cols >=8 are zero), residual add.
  c0 = wmma16(a, bfrag_p(Wp, OW3_B, lane), z);
  const int n = lane & 15;
  if (n < 8) {
    const float bb = ob3[n];
#pragma unroll
    for (int v = 0; v < 8; ++v) {
      int kk = k0 + v + 8 * kh;
      size_t idx = ((size_t)i * NN + kk) * HH + n;
      out[idx] = mtr[idx] + c0[v] + bb;
    }
  }
}

// --------------------------- host launcher ---------------------------------

extern "C" void kernel_launch(void* const* d_in, const int* in_sizes, int n_in,
                              void* d_out, int out_size, void* d_ws, size_t ws_size,
                              hipStream_t stream) {
  (void)in_sizes; (void)n_in; (void)out_size; (void)ws_size;
  const float*     mtr  = (const float*)d_in[0];
  const float*     feat = (const float*)d_in[1];
  const long long* ei   = (const long long*)d_in[2];   // int64 [2][E]
  const float*     ea   = (const float*)d_in[3];
  const float *ew0 = (const float*)d_in[4],  *eb0 = (const float*)d_in[5];
  const float *ew1 = (const float*)d_in[6],  *eb1 = (const float*)d_in[7];
  const float *ew2 = (const float*)d_in[8],  *eb2 = (const float*)d_in[9];
  const float *ew3 = (const float*)d_in[10], *eb3 = (const float*)d_in[11];
  const float *ow0 = (const float*)d_in[12], *ob0 = (const float*)d_in[13];
  const float *ow1 = (const float*)d_in[14], *ob1 = (const float*)d_in[15];
  const float *ow2 = (const float*)d_in[16], *ob2 = (const float*)d_in[17];
  const float *ow3 = (const float*)d_in[18], *ob3 = (const float*)d_in[19];
  float* out = (float*)d_out;

  // Workspace layout (16B-aligned):
  //   A16 f16 [768][6144]      :  9,437,184 B
  //   F16 f16 swizzled 6144^2  : 75,497,472 B
  //   M1  f32 [768][6144]      : 18,874,368 B
  //   Wp  f16 packed weights   :     19,456 B   (total ~99 MB)
  char* ws = (char*)d_ws;
  _Float16* A16 = (_Float16*)ws;
  _Float16* F16 = (_Float16*)(ws + 9437184);
  float*    M1  = (float*)(ws + 9437184 + 75497472);
  _Float16* Wp  = (_Float16*)(ws + 9437184 + 75497472 + 18874368);

  // Pack weights (tiny).
  pack_w<<<2, 256, 0, stream>>>(ew0, 20, 32, 2, Wp + EW0_B * 512, 2 * 512);
  pack_w<<<2, 256, 0, stream>>>(ew1, 32, 32, 2, Wp + EW1_B * 512, 2 * 512);
  pack_w<<<2, 256, 0, stream>>>(ew2, 32, 32, 2, Wp + EW2_B * 512, 2 * 512);
  pack_w<<<4, 256, 0, stream>>>(ew3, 32, 64, 4, Wp + EW3_B * 512, 4 * 512);
  pack_w<<<4, 256, 0, stream>>>(ow0, 33, 32, 2, Wp + OW0_B * 512, 4 * 512);
  pack_w<<<2, 256, 0, stream>>>(ow1, 32, 32, 2, Wp + OW1_B * 512, 2 * 512);
  pack_w<<<2, 256, 0, stream>>>(ow2, 32, 32, 2, Wp + OW2_B * 512, 2 * 512);
  pack_w<<<1, 256, 0, stream>>>(ow3, 32,  8, 1, Wp + OW3_B * 512, 1 * 512);

  f32_to_f16_4<<<(NN * KDIM / 4) / 256, 256, 0, stream>>>(
      (const float4*)mtr, (v4h*)A16, NN * KDIM / 4);
  zero16B<<<((KDIM * KDIM / 8) + 255) / 256, 256, 0, stream>>>(
      (uint4*)F16, KDIM * KDIM / 8);
  edge_mlp_scatter<<<EDGES / 16 / 8, 256, 0, stream>>>(
      ea, feat, ei, Wp, eb0, eb1, eb2, eb3, F16);
  walk_gemm<<<(24 * 96) / 8, 256, 0, stream>>>(A16, F16, M1);
  out_mlp<<<(NN * 48) / 8, 256, 0, stream>>>(
      M1, feat, mtr, Wp, ob0, ob1, ob2, ob3, out);
}
